// SAGELayer_6545530159693
// MI455X (gfx1250) — matrix operand order repaired and checked by reading, not compiled, not run on hardware
//
#include <hip/hip_runtime.h>
#include <stdint.h>

// GraphSAGE layer for MI455X (gfx1250, wave32, WMMA).
// Strategy: exploit linearity of W_msg -> aggregate features first (pure
// bandwidth), then two small node-level GEMMs with v_wmma_f32_16x16x32_bf16.

#define NNODES 50000
#define NEDGES 800000
#define KDIM   256      // NDIM_IN + EDIM (and NDIM_IN + NDIM_OUT)
#define NDIM   128      // feature dim / out dim

typedef __attribute__((ext_vector_type(16))) __bf16 v16bf;
typedef __attribute__((ext_vector_type(8)))  __bf16 v8bf;
typedef __attribute__((ext_vector_type(8)))  float  v8f;

static __device__ __forceinline__ __bf16 f32_to_bf16(float f) {
  union { float f; uint32_t u; } in; in.f = f;
  uint32_t u = in.u;
  uint32_t r = u + 0x7FFFu + ((u >> 16) & 1u);          // round-to-nearest-even
  if ((u & 0x7F800000u) == 0x7F800000u) r = u;          // preserve inf/nan
  union { uint16_t s; __bf16 b; } out; out.s = (uint16_t)(r >> 16);
  return out.b;
}

// ---------------- prep kernels ----------------

__global__ void k_zero_i32(int* __restrict__ a, int n) {
  int i = blockIdx.x * blockDim.x + threadIdx.x;
  if (i < n) a[i] = 0;
}

__global__ void k_convert_bf16(const float* __restrict__ in, __bf16* __restrict__ out, int n) {
  int i = blockIdx.x * blockDim.x + threadIdx.x;
  if (i < n) out[i] = f32_to_bf16(in[i]);
}

// nfeats [N][128] f32 -> appin[:, 0:128] bf16 (row stride KDIM)
__global__ void k_nfeats_to_appin(const float* __restrict__ nf, __bf16* __restrict__ appin) {
  int i = blockIdx.x * blockDim.x + threadIdx.x;
  if (i < NNODES * NDIM) {
    int r = i >> 7, c = i & 127;
    appin[r * KDIM + c] = f32_to_bf16(nf[i]);
  }
}

// ---------------- CSR build (by dst) ----------------

__global__ void k_count(const int* __restrict__ dst, int* __restrict__ counts) {
  int e = blockIdx.x * blockDim.x + threadIdx.x;
  if (e < NEDGES) atomicAdd(&counts[dst[e]], 1);
}

// single-block exclusive scan over 50k counters (chunked Hillis-Steele in LDS)
__global__ void k_scan(const int* __restrict__ counts, int* __restrict__ offsets) {
  __shared__ int buf[256];
  __shared__ int carry;
  if (threadIdx.x == 0) { carry = 0; offsets[0] = 0; }
  __syncthreads();
  for (int base = 0; base < NNODES; base += 256) {
    int i = base + (int)threadIdx.x;
    buf[threadIdx.x] = (i < NNODES) ? counts[i] : 0;
    __syncthreads();
    for (int off = 1; off < 256; off <<= 1) {
      int add = (threadIdx.x >= (unsigned)off) ? buf[threadIdx.x - off] : 0;
      __syncthreads();
      buf[threadIdx.x] += add;
      __syncthreads();
    }
    if (i < NNODES) offsets[i + 1] = carry + buf[threadIdx.x];
    __syncthreads();
    if (threadIdx.x == 0) carry += buf[255];
    __syncthreads();
  }
}

__global__ void k_fill(const int* __restrict__ dst, const int* __restrict__ offsets,
                       int* __restrict__ cursor, int* __restrict__ eidx) {
  int e = blockIdx.x * blockDim.x + threadIdx.x;
  if (e < NEDGES) {
    int d = dst[e];
    int pos = atomicAdd(&cursor[d], 1);
    eidx[offsets[d] + pos] = e;
  }
}

// ---------------- aggregation (pure bandwidth) ----------------
// one 128-thread block per node; thread t owns channel t.
// agg[v] = [ mean(nfeats[src]) | mean(efeats) ]  (bf16, zeros if deg==0)

__global__ void k_aggregate(const float* __restrict__ nfeats, const float* __restrict__ efeats,
                            const int* __restrict__ src, const int* __restrict__ offsets,
                            const int* __restrict__ eidx, __bf16* __restrict__ agg) {
  int v = blockIdx.x;
  int t = threadIdx.x;                 // 0..127
  int beg = offsets[v], end = offsets[v + 1];
  float ah = 0.f, ae = 0.f;
  for (int i = beg; i < end; ++i) {
    int e = eidx[i];
    int s = src[e];
    ah += nfeats[(size_t)s * NDIM + t];
    ae += efeats[(size_t)e * NDIM + t];
  }
  int deg = end - beg;
  float sc = (deg > 0) ? 1.0f / (float)deg : 0.0f;
  agg[(size_t)v * KDIM + t]        = f32_to_bf16(ah * sc);
  agg[(size_t)v * KDIM + NDIM + t] = f32_to_bf16(ae * sc);
}

// ---------------- WMMA GEMM: [M x 256] x [256 x 128] ----------------
// 256 threads = 8 waves; wave w owns N-tile n0 = 16*w, holds its 8 weight
// fragments (K=256) in registers for the whole kernel, loops over M-tiles.
// MODE 0: h_neigh = A*W_msg^T + b_msg, masked to 0 where deg==0, written as
//         bf16 into appin[:, 128:256].
// MODE 1: out = relu(A*W_apply^T + b_apply) written f32 to d_out.

#define MTILES_TOTAL (NNODES / 16)   // 3125 (exact)
#define MTILES_PER_BLOCK 4

template <int MODE>
__global__ __launch_bounds__(256) void k_gemm(
    const __bf16* __restrict__ A,     // [M][256] bf16
    const __bf16* __restrict__ W,     // [128][256] bf16 (row n = output channel)
    const float*  __restrict__ bias,  // [128]
    const int*    __restrict__ offsets,
    __bf16*       __restrict__ out_bf,
    float*        __restrict__ out_f) {
  const int wave = threadIdx.x >> 5;
  const int lane = threadIdx.x & 31;
  const int n0   = wave * 16;
  const int lrow = lane & 15;
  const int g    = lane >> 4;

  // B fragments: B[kk][nn] = W[n0+nn][k0+kk]; lane holds N=lrow, K=k0+16g+i
  v16bf bfrag[8];
#pragma unroll
  for (int s = 0; s < 8; ++s) {
    const __bf16* p = W + (size_t)(n0 + lrow) * KDIM + s * 32 + g * 16;
    bfrag[s] = *(const v16bf*)p;      // 32B contiguous -> 2x global_load_b128
  }
  const float bval = bias[n0 + lrow];

  for (int tt = 0; tt < MTILES_PER_BLOCK; ++tt) {
    int mt = blockIdx.x * MTILES_PER_BLOCK + tt;
    if (mt >= MTILES_TOTAL) break;    // uniform across block -> EXEC stays full
    int m0 = mt * 16;

    v8f c = {};
#pragma unroll
    for (int s = 0; s < 8; ++s) {
      // A fragment: lane M=lrow; K = k0+16g+i for elems 0..15 (matches B)
      const __bf16* pa = A + (size_t)(m0 + lrow) * KDIM + s * 32 + g * 16;
      v8bf lo = *(const v8bf*)pa;
      v8bf hi = *(const v8bf*)(pa + 8);
      v16bf a;
#pragma unroll
      for (int i = 0; i < 8; ++i) { a[i] = lo[i]; a[i + 8] = hi[i]; }
#if defined(__gfx1250__)
      c = __builtin_amdgcn_wmma_f32_16x16x32_bf16(
          /*neg_a=*/false, a, /*neg_b=*/false, bfrag[s],
          /*c_mod=*/(short)0, c, /*reuse_a=*/false, /*reuse_b=*/false);
#endif
    }

    // D layout: element r -> row m0 + r + 8g, col n0 + lrow
    int col = n0 + lrow;
#pragma unroll
    for (int r = 0; r < 8; ++r) {
      int m = m0 + r + 8 * g;
      float val = c[r] + bval;
      if (MODE == 0) {
        int deg = offsets[m + 1] - offsets[m];
        if (deg <= 0) val = 0.0f;                      // mean over empty set is 0
        out_bf[(size_t)m * KDIM + NDIM + col] = f32_to_bf16(val);
      } else {
        out_f[(size_t)m * NDIM + col] = val > 0.0f ? val : 0.0f;   // relu
      }
    }
  }
}

// ---------------- launch ----------------

extern "C" void kernel_launch(void* const* d_in, const int* in_sizes, int n_in,
                              void* d_out, int out_size, void* d_ws, size_t ws_size,
                              hipStream_t stream) {
  (void)in_sizes; (void)n_in; (void)out_size; (void)ws_size;
  const float* nfeats = (const float*)d_in[0];
  const float* efeats = (const float*)d_in[1];
  const int*   src    = (const int*)d_in[2];
  const int*   dst    = (const int*)d_in[3];
  const float* Wmsg   = (const float*)d_in[4];
  const float* bmsg   = (const float*)d_in[5];
  const float* Wapp   = (const float*)d_in[6];
  const float* bapp   = (const float*)d_in[7];
  float* out = (float*)d_out;

  // carve workspace (256B-aligned regions)
  char* p = (char*)d_ws;
  auto carve = [&](size_t bytes) -> char* {
    char* r = p;
    p += (bytes + 255) & ~(size_t)255;
    return r;
  };
  int*    counts  = (int*)carve((size_t)NNODES * 4);
  int*    cursor  = (int*)carve((size_t)NNODES * 4);
  int*    offsets = (int*)carve((size_t)(NNODES + 1) * 4);
  int*    eidx    = (int*)carve((size_t)NEDGES * 4);
  __bf16* Wmsg_bf = (__bf16*)carve((size_t)NDIM * KDIM * 2);
  __bf16* Wapp_bf = (__bf16*)carve((size_t)NDIM * KDIM * 2);
  __bf16* agg     = (__bf16*)carve((size_t)NNODES * KDIM * 2);
  __bf16* appin   = (__bf16*)carve((size_t)NNODES * KDIM * 2);

  const int T = 256;
  k_zero_i32<<<(NNODES + T - 1) / T, T, 0, stream>>>(counts, NNODES);
  k_zero_i32<<<(NNODES + T - 1) / T, T, 0, stream>>>(cursor, NNODES);

  const int wElems = NDIM * KDIM;  // 32768
  k_convert_bf16<<<(wElems + T - 1) / T, T, 0, stream>>>(Wmsg, Wmsg_bf, wElems);
  k_convert_bf16<<<(wElems + T - 1) / T, T, 0, stream>>>(Wapp, Wapp_bf, wElems);
  k_nfeats_to_appin<<<(NNODES * NDIM + T - 1) / T, T, 0, stream>>>(nfeats, appin);

  k_count<<<(NEDGES + T - 1) / T, T, 0, stream>>>(dst, counts);
  k_scan<<<1, 256, 0, stream>>>(counts, offsets);
  k_fill<<<(NEDGES + T - 1) / T, T, 0, stream>>>(dst, offsets, cursor, eidx);

  k_aggregate<<<NNODES, 128, 0, stream>>>(nfeats, efeats, src, offsets, eidx, agg);

  const int gemmBlocks = (MTILES_TOTAL + MTILES_PER_BLOCK - 1) / MTILES_PER_BLOCK;
  k_gemm<0><<<gemmBlocks, 256, 0, stream>>>(agg,   Wmsg_bf, bmsg, offsets, appin, nullptr);
  k_gemm<1><<<gemmBlocks, 256, 0, stream>>>(appin, Wapp_bf, bapp, offsets, nullptr, out);
}
